// SurfCrossModalityDecoder_44014824849613
// MI455X (gfx1250) — matrix operand order repaired and checked by reading, not compiled, not run on hardware
//
#include <hip/hip_runtime.h>
#include <hip/hip_bf16.h>

// ---------------------------------------------------------------------------
// SurfCrossModalityDecoder for MI455X (gfx1250, wave32, WMMA)
// Dominant cost: edge MLP [E=800k,256]@[256,256] ~105 GFLOP -> f16 WMMA with
// the B operand streamed through LDS by GLOBAL_LOAD_ASYNC_TO_LDS_B128
// (ASYNCcnt-tracked, 3-tile issue-ahead ring buffer). Dual WMMA accumulator
// chains break the serial D->C dependence so LDS latency overlaps compute.
// ---------------------------------------------------------------------------

typedef __attribute__((ext_vector_type(16))) _Float16 v16h;
typedef __attribute__((ext_vector_type(8)))  _Float16 v8h;
typedef __attribute__((ext_vector_type(4)))  _Float16 v4h;
typedef __attribute__((ext_vector_type(8)))  float    v8f;

#define C_IN   128
#define C_HID  256
#define C_OUT  128
#define SEGSTRIDE 132          // [attr, 128 feats, count, pad, pad]
#define EPW    64              // edges per workgroup (4 M-tiles, one per wave)
#define APAD   264             // LDS half stride for K=256 rows (16B aligned)
#define A1PAD  136             // LDS half stride for K=128 rows
#define NBUF   4               // async B ring-buffer depth
#define BTILE_BYTES (16 * 264 * 2)   // one staged B tile (16 cols x 264 halves)

static __device__ __forceinline__ v8f wmma16(v16h a, v16h b, v8f c) {
  // D = A(16x32 f16) * B(32x16 f16) + C(16x16 f32)
  return __builtin_amdgcn_wmma_f32_16x16x32_f16(
      /*neg_a=*/false, a, /*neg_b=*/false, b,
      /*c_mod=*/(short)0, c, /*reuse_a=*/false, /*reuse_b=*/false);
}

// Load A fragment (16x32 f16 tile slice for this lane) from an LDS row.
// Contiguous K order used here (B loads match): lanes 0-15 -> row = lane, halves 0..15 = K(kb..kb+15);
// lanes 16-31 same row, +16 K offset.
static __device__ __forceinline__ v16h load_afrag(const _Float16* rowK,
                                                  int kbase, int kg) {
  union { v16h v; v8h h[2]; } a;
  const _Float16* p = rowK + kbase + kg * 16;
  a.h[0] = *(const v8h*)(p);
  a.h[1] = *(const v8h*)(p + 8);
  return a.v;
}

// Async-copy one 16B chunk global -> LDS (per lane), tracked by ASYNCcnt.
static __device__ __forceinline__ void async_g2l_b128(unsigned lds_off,
                                                      const void* gaddr) {
  asm volatile("global_load_async_to_lds_b128 %0, %1, off"
               :: "v"(lds_off), "v"((unsigned long long)(uintptr_t)gaddr)
               : "memory");
}

// ---------------------------------------------------------------------------
// Kernel 0: convert weights fp32 -> f16, transposed to K-major per out column
// so a lane's B-fragment (16 contiguous K halves) is one 32B load.
// ---------------------------------------------------------------------------
__global__ __launch_bounds__(256) void prep_weights(
    const float* __restrict__ We1, const float* __restrict__ Wa1,
    const float* __restrict__ Wa2,
    _Float16* __restrict__ We1h, _Float16* __restrict__ Wa1h,
    _Float16* __restrict__ Wa2h) {
  int t = blockIdx.x * 256 + threadIdx.x;
  if (t < 65536) {                       // We1: [256,256] -> [n][k]
    int n = t >> 8, k = t & 255;
    We1h[(size_t)n * 256 + k] = (_Float16)We1[(size_t)k * 256 + n];
  } else if (t < 98304) {                // Wa1 rows 1..128: [129,256] -> [n][k]
    int u = t - 65536; int n = u >> 7, k = u & 127;
    Wa1h[(size_t)n * 128 + k] = (_Float16)Wa1[(size_t)(k + 1) * 256 + n];
  } else if (t < 131072) {               // Wa2: [256,128] -> [n][k]
    int u = t - 98304; int n = u >> 8, k = u & 255;
    Wa2h[(size_t)n * 256 + k] = (_Float16)Wa2[(size_t)k * 128 + n];
  }
}

// ---------------------------------------------------------------------------
// Kernel 1: zero the segment-sum workspace
// ---------------------------------------------------------------------------
__global__ __launch_bounds__(256) void zero_seg(float* __restrict__ p,
                                                size_t n) {
  size_t i = (size_t)blockIdx.x * 256 + threadIdx.x;
  if (i < n) p[i] = 0.0f;
}

// ---------------------------------------------------------------------------
// Kernel 2: scatter-sum of [edge_attr, z_src] over dst (+ counts).
// One wave per edge; 104M f32 atomics land in the 192MB L2.
// ---------------------------------------------------------------------------
__global__ __launch_bounds__(128) void scatter_kernel(
    const float* __restrict__ z, const int* __restrict__ ei,
    const float* __restrict__ ea, float* __restrict__ seg, int E_) {
  int e = blockIdx.x * 4 + (threadIdx.x >> 5);
  int lane = threadIdx.x & 31;
  if (e >= E_) return;
  int src = ei[e];
  int dst = ei[E_ + e];
  float* row = seg + (size_t)dst * SEGSTRIDE;
  const float* zs = z + (size_t)src * C_IN;
#pragma unroll
  for (int i = 0; i < 4; ++i) {
    int c = lane + 32 * i;
    atomicAdd(&row[1 + c], zs[c]);
  }
  if (lane == 0) {
    atomicAdd(&row[0], ea[e]);
    atomicAdd(&row[C_IN + 1], 1.0f);   // count at col 129
  }
}

// ---------------------------------------------------------------------------
// Kernel 3: edge decoder MLP.
// WG = 128 threads (4 waves), 64 edges. Wave w owns M-tile w (16 edges):
// its A fragments (8 x v16h = 64 VGPRs) are loaded once from LDS and reused
// for all 16 column tiles. B column tiles (8KB each) are staged into a
// 4-deep LDS ring buffer with global_load_async_to_lds_b128, issue-ahead 3,
// synchronized via s_wait_asynccnt + one barrier per tile. The K loop uses
// two independent WMMA accumulator chains (even/odd K-steps) so B-fragment
// LDS latency hides under the other chain's WMMA. Layer 2 accumulates
// relu(h)*We2 in registers, then one LDS deposit + row sum.
// ---------------------------------------------------------------------------
__global__ __launch_bounds__(128) void edge_mlp_kernel(
    const float* __restrict__ z, const int* __restrict__ ei,
    const _Float16* __restrict__ We1h, const float* __restrict__ be1,
    const float* __restrict__ We2, const float* __restrict__ be2,
    float* __restrict__ erec, int E_) {
  __shared__ _Float16 A[EPW][APAD];             // 64 x 256 f16 edge_input (~33KB)
  __shared__ _Float16 Bbuf[NBUF][16][APAD];     // async B ring buffer   (~33KB)
  __shared__ float R[EPW][17];                  // per-(row,lane) partials (~4KB)

  const int t  = threadIdx.x;
  const int wg = blockIdx.x;

  const unsigned bb = (unsigned)(uintptr_t)(&Bbuf[0][0][0]);  // LDS byte offset
  const char* bsrc = (const char*)We1h;

  // ---- prologue: async-issue B tiles 0..2 into ring buffers ------------
  // One tile = 16 weight columns x 512B = 8KB = 512 x 16B chunks;
  // 128 threads x 4 chunks. LDS rows use the padded 264-half stride.
#pragma unroll
  for (int pf = 0; pf < 3; ++pf) {
#pragma unroll
    for (int q = 0; q < 4; ++q) {
      int l = q * 128 + t;                       // 0..511
      unsigned lds = bb + (unsigned)(pf * BTILE_BYTES + (l >> 5) * (APAD * 2) +
                                     (l & 31) * 16);
      async_g2l_b128(lds, bsrc + (size_t)pf * 8192 + (size_t)l * 16);
    }
  }

  // ---- gather z_src || z_dst, convert to f16 in LDS --------------------
  {
    int e    = t >> 1;                  // 0..63 edge within tile
    int hsel = t & 1;                   // 0 -> src half, 1 -> dst half
    int eg   = wg * EPW + e;
    int egc  = eg < E_ ? eg : E_ - 1;
    int node = hsel ? ei[E_ + egc] : ei[egc];
    const float4* zp = (const float4*)(z + (size_t)node * C_IN);
    _Float16* dst = &A[e][hsel * C_IN];
#pragma unroll
    for (int i = 0; i < 32; ++i) {
      float4 v = zp[i];
      v4h h;
      h.x = (_Float16)v.x; h.y = (_Float16)v.y;
      h.z = (_Float16)v.z; h.w = (_Float16)v.w;
      *(v4h*)(dst + 4 * i) = h;
    }
  }
  __syncthreads();

  const int lane = t & 31;
  const int wave = t >> 5;              // owns M-tile `wave`
  const int nlo  = lane & 15;           // A-row / B-col / D-col index
  const int kg   = lane >> 4;           // lane half (K group)

  // A fragments for this wave's 16 edges, all 8 K-steps (64 VGPRs).
  v16h Af[8];
  {
    const _Float16* arow = &A[wave * 16 + nlo][0];
#pragma unroll
    for (int s = 0; s < 8; ++s) Af[s] = load_afrag(arow, s * 32, kg);
  }

  float racc[8];
#pragma unroll
  for (int i = 0; i < 8; ++i) racc[i] = 0.0f;

#pragma unroll 1
  for (int ct = 0; ct < 16; ++ct) {
    // Wait for this tile's own async issues (4 per tile; up to 3 tiles out).
    if (ct < 14)       asm volatile("s_wait_asynccnt 0x8" ::: "memory");
    else if (ct == 14) asm volatile("s_wait_asynccnt 0x4" ::: "memory");
    else               asm volatile("s_wait_asynccnt 0x0" ::: "memory");
    __syncthreads();   // tile ct visible to all waves; all done with ct-1

    // Issue-ahead: tile ct+3 into buffer (ct+3)&3 (last read at iter ct-1).
    if (ct + 3 < 16) {
      int T = ct + 3;
#pragma unroll
      for (int q = 0; q < 4; ++q) {
        int l = q * 128 + t;
        unsigned lds = bb + (unsigned)((T & 3) * BTILE_BYTES +
                                       (l >> 5) * (APAD * 2) + (l & 31) * 16);
        async_g2l_b128(lds, bsrc + (size_t)T * 8192 + (size_t)l * 16);
      }
    }

    // ---- compute this wave's M-tile against column tile ct -------------
    const int ncol = ct * 16 + nlo;             // hidden column 0..255
    const float bias = be1[ncol];
    const float w2   = We2[ncol];
    const _Float16* brow = &Bbuf[ct & 3][nlo][0];

    // Two independent accumulator chains (even/odd K-steps): breaks the
    // serial WMMA D->C dependence so each chain's B load latency hides
    // under the other chain's WMMA execution.
    v8f c0, c1;
#pragma unroll
    for (int i = 0; i < 8; ++i) { c0[i] = bias; c1[i] = 0.0f; }
#pragma unroll
    for (int s = 0; s < 4; ++s) {
      v16h b0 = *(const v16h*)(brow + (2 * s)     * 32 + kg * 16);
      v16h b1 = *(const v16h*)(brow + (2 * s + 1) * 32 + kg * 16);
      c0 = wmma16(Af[2 * s],     b0, c0);
      c1 = wmma16(Af[2 * s + 1], b1, c1);
    }
#pragma unroll
    for (int i = 0; i < 8; ++i)
      racc[i] += fmaxf(c0[i] + c1[i], 0.0f) * w2;
  }

  // ---- deposit partials (distinct addresses, padded stride) ------------
#pragma unroll
  for (int i = 0; i < 8; ++i)
    R[wave * 16 + kg * 8 + i][nlo] = racc[i];
  __syncthreads();

  // ---- final: one thread per edge sums 16 lane-partials + be2 ----------
  if (t < EPW) {
    float s = be2[0];
#pragma unroll
    for (int j = 0; j < 16; ++j) s += R[t][j];
    int eg = wg * EPW + t;
    if (eg < E_) erec[eg] = s;
  }
}

// ---------------------------------------------------------------------------
// Kernel 4: node attribute decoder MLP on scatter-mean context.
// K=129 split: K=128 via WMMA + rank-1 (ctx0 * Wa1 row 0) folded into C init.
// ---------------------------------------------------------------------------
__global__ __launch_bounds__(128) void node_mlp_kernel(
    const float* __restrict__ seg, const _Float16* __restrict__ Wa1h,
    const float* __restrict__ Wa1, const float* __restrict__ ba1,
    const _Float16* __restrict__ Wa2h, const float* __restrict__ ba2,
    float* __restrict__ xrec, int N_) {
  __shared__ _Float16 A1[16][A1PAD];    // struct_ctx[:,1:129] f16
  __shared__ _Float16 H[16][APAD];      // relu hidden, f16
  __shared__ float ctx0[16];            // struct_ctx[:,0] fp32

  const int t     = threadIdx.x;
  const int nbase = blockIdx.x * 16;

  // ---- scatter-mean normalize + f16 convert ----------------------------
  {
    int m = t >> 3, p = t & 7;
    int node = nbase + m;
    if (node >= N_) node = N_ - 1;
    const float* row = seg + (size_t)node * SEGSTRIDE;
    float cnt = row[C_IN + 1];
    float inv = cnt > 0.0f ? 1.0f / cnt : 0.0f;
    if (p == 0) ctx0[m] = row[0] * inv;
#pragma unroll
    for (int i = 0; i < 16; ++i) {
      int cIdx = p * 16 + i;            // 0..127
      A1[m][cIdx] = (_Float16)(row[1 + cIdx] * inv);
    }
  }
  __syncthreads();

  const int lane = t & 31;
  const int wave = t >> 5;
  const int nlo  = lane & 15;
  const int kg   = lane >> 4;

  // ---- layer 1: [16,128] @ Wa1h(128xC_HID) + rank-1 + bias, ReLU -------
#pragma unroll 1
  for (int ct = 0; ct < 4; ++ct) {
    const int ncol = (wave * 4 + ct) * 16 + nlo;   // 0..255
    const float w0 = Wa1[ncol];                    // Wa1 row 0 (attr column)
    const float b  = ba1[ncol];
    const _Float16* arow = &A1[nlo][0];
    v8f c0, c1;
#pragma unroll
    for (int i = 0; i < 8; ++i) { c0[i] = b + ctx0[kg * 8 + i] * w0; c1[i] = 0.0f; }
#pragma unroll
    for (int s = 0; s < 2; ++s) {
      v16h b0 = *(const v16h*)(Wa1h + (size_t)ncol * 128 + (2 * s)     * 32 + kg * 16);
      v16h b1 = *(const v16h*)(Wa1h + (size_t)ncol * 128 + (2 * s + 1) * 32 + kg * 16);
      c0 = wmma16(load_afrag(arow, (2 * s)     * 32, kg), b0, c0);
      c1 = wmma16(load_afrag(arow, (2 * s + 1) * 32, kg), b1, c1);
    }
#pragma unroll
    for (int i = 0; i < 8; ++i)
      H[kg * 8 + i][ncol] = (_Float16)fmaxf(c0[i] + c1[i], 0.0f);
  }
  __syncthreads();

  // ---- layer 2: [16,256] @ Wa2h(256xC_OUT) + ba2 -----------------------
#pragma unroll 1
  for (int ot = 0; ot < 2; ++ot) {
    const int ocol = (wave * 2 + ot) * 16 + nlo;   // 0..127
    const float b2 = ba2[ocol];
    const _Float16* arow = &H[nlo][0];
    v8f c0, c1;
#pragma unroll
    for (int i = 0; i < 8; ++i) { c0[i] = b2; c1[i] = 0.0f; }
#pragma unroll
    for (int s = 0; s < 4; ++s) {
      v16h b0 = *(const v16h*)(Wa2h + (size_t)ocol * 256 + (2 * s)     * 32 + kg * 16);
      v16h b1 = *(const v16h*)(Wa2h + (size_t)ocol * 256 + (2 * s + 1) * 32 + kg * 16);
      c0 = wmma16(load_afrag(arow, (2 * s)     * 32, kg), b0, c0);
      c1 = wmma16(load_afrag(arow, (2 * s + 1) * 32, kg), b1, c1);
    }
#pragma unroll
    for (int i = 0; i < 8; ++i) {
      int node = nbase + kg * 8 + i;
      if (node < N_) xrec[(size_t)node * C_OUT + ocol] = c0[i] + c1[i];
    }
  }
}

// ---------------------------------------------------------------------------
extern "C" void kernel_launch(void* const* d_in, const int* in_sizes, int n_in,
                              void* d_out, int out_size, void* d_ws,
                              size_t ws_size, hipStream_t stream) {
  (void)n_in; (void)out_size; (void)ws_size;
  const float* z   = (const float*)d_in[0];
  const int*   ei  = (const int*)d_in[1];    // edge_index [2,E] (int32)
  const float* ea  = (const float*)d_in[2];
  const float* Wa1 = (const float*)d_in[3];
  const float* ba1 = (const float*)d_in[4];
  const float* Wa2 = (const float*)d_in[5];
  const float* ba2 = (const float*)d_in[6];
  const float* We1 = (const float*)d_in[7];
  const float* be1 = (const float*)d_in[8];
  const float* We2 = (const float*)d_in[9];
  const float* be2 = (const float*)d_in[10];

  const int N = in_sizes[0] / C_IN;
  const int E = in_sizes[1] / 2;

  // workspace layout (all offsets 256B-aligned)
  char* ws = (char*)d_ws;
  _Float16* We1h = (_Float16*)(ws + 0);        // 256*256 f16 = 128KB
  _Float16* Wa1h = (_Float16*)(ws + 131072);   // 128*256 f16 =  64KB
  _Float16* Wa2h = (_Float16*)(ws + 196608);   // 256*128 f16 =  64KB
  float*    seg  = (float*)(ws + 262144);      // N*132 f32   ~26.4MB

  float* xrec = (float*)d_out;                 // [N,128]
  float* erec = (float*)d_out + (size_t)N * C_OUT;  // [E,1]

  prep_weights<<<512, 256, 0, stream>>>(We1, Wa1, Wa2, We1h, Wa1h, Wa2h);

  size_t segN = (size_t)N * SEGSTRIDE;
  zero_seg<<<(unsigned)((segN + 255) / 256), 256, 0, stream>>>(seg, segN);

  scatter_kernel<<<(E + 3) / 4, 128, 0, stream>>>(z, ei, ea, seg, E);

  edge_mlp_kernel<<<(E + EPW - 1) / EPW, 128, 0, stream>>>(
      z, ei, We1h, be1, We2, be2, erec, E);

  node_mlp_kernel<<<(N + 15) / 16, 128, 0, stream>>>(
      seg, Wa1h, Wa1, ba1, Wa2h, ba2, xrec, N);
}
